// ScaledDotProductAttention_2078764171488
// MI455X (gfx1250) — matrix operand order repaired and checked by reading, not compiled, not run on hardware
//
#include <hip/hip_runtime.h>
#include <hip/hip_bf16.h>
#include <hip/hip_fp16.h>

typedef __attribute__((ext_vector_type(16))) _Float16 v16h;
typedef __attribute__((ext_vector_type(8)))  float    v8f;
typedef __attribute__((ext_vector_type(4)))  float    f32x4;

#define ATT_B 2
#define ATT_H 16
#define ATT_S 2048
#define ATT_D 64

// scale = 1/sqrt(64) folded with log2(e) so we can use v_exp_f32 (exp2)
#define CSC 0.18033688011112042592f /* 0.125 * 1.4426950408889634 */

__device__ __forceinline__ float fexp2(float x) {
#if __has_builtin(__builtin_amdgcn_exp2f)
  return __builtin_amdgcn_exp2f(x);
#else
  return exp2f(x);
#endif
}

// A-matrix (16x32 f16) fragment for one 32-wide d/k range starting at d0.
// Lane l holds row M = l&15; elem i -> k = d0 + (i<8 ? 8*half+i : 16+8*half+(i-8)).
__device__ __forceinline__ v16h mk_a_f32(const float* __restrict__ row, int d0, int half) {
  const float* p0 = row + d0 + (half << 3);
  const float* p1 = p0 + 16;
  v16h a;
#pragma unroll
  for (int i = 0; i < 8; ++i) {
    a[i]     = (_Float16)p0[i];
    a[i + 8] = (_Float16)p1[i];
  }
  return a;
}

// B-matrix (32x16 f16) fragment from a K row (row-major, D contiguous).
// Lane l holds col N = l&15; elem i -> k = d0 + (i<8 ? 8*half+i : 16+8*half+(i-8)).
__device__ __forceinline__ v16h mk_b_krow(const float* __restrict__ krow, int d0, int half) {
  const float* p0 = krow + d0 + (half << 3);
  const float* p1 = p0 + 16;
  v16h b;
#pragma unroll
  for (int i = 0; i < 8; ++i) {
    b[i]     = (_Float16)p0[i];
    b[i + 8] = (_Float16)p1[i];
  }
  return b;
}

// One 16x16 score tile: S = Q(16x64) * K^T(64x16), K-dim split into 2 WMMAs.
__device__ __forceinline__ v8f qk_tile(v16h a0, v16h a1, const float* __restrict__ Kbh,
                                       int k0, int lo, int half) {
  const float* krow = Kbh + (size_t)(k0 + lo) * ATT_D;
  v16h b0 = mk_b_krow(krow, 0, half);
  v16h b1 = mk_b_krow(krow, 32, half);
  v8f c = {};
  c = __builtin_amdgcn_wmma_f32_16x16x32_f16(false, a0, false, b0, (short)0, c, false, false);
  c = __builtin_amdgcn_wmma_f32_16x16x32_f16(false, a1, false, b1, (short)0, c, false, false);
  return c;
}

// One wave (32 threads) per (b*h, 16-row q tile).
__global__ void __launch_bounds__(32)
attn_fwd_kernel(const float* __restrict__ Q, const float* __restrict__ K,
                const float* __restrict__ V, float* __restrict__ O,
                float* __restrict__ W) {
  const int lane = threadIdx.x;        // 0..31 (wave32)
  const int lo   = lane & 15;
  const int half = lane >> 4;

  const int qt = blockIdx.x & 127;     // q-tile index (S/16 = 128)
  const int bh = blockIdx.x >> 7;      // fused batch*head (0..31)
  const int q0 = qt << 4;

  const float* Qbh = Q + (size_t)bh * ATT_S * ATT_D;
  const float* Kbh = K + (size_t)bh * ATT_S * ATT_D;
  const float* Vbh = V + (size_t)bh * ATT_S * ATT_D;
  float*       Obh = O + (size_t)bh * ATT_S * ATT_D;
  float*       Wbh = W + (size_t)bh * ATT_S * ATT_S;

  // Q tile in A-layout (row M = lo)
  const float* qrow = Qbh + (size_t)(q0 + lo) * ATT_D;
  const v16h a0 = mk_a_f32(qrow, 0, half);
  const v16h a1 = mk_a_f32(qrow, 32, half);

  // ---------------- Pass A: online softmax statistics (row max / row sum) ----
  // C-layout: lane holds col N = lo of rows M = v + 8*half.
  float m[8], sm[8];
#pragma unroll
  for (int v = 0; v < 8; ++v) { m[v] = -__builtin_inff(); sm[v] = 0.0f; }

  for (int j = 0; j <= qt; ++j) {
    const int k0 = j << 4;
    if (j < qt)  // prefetch next K tile row for this lane (always in-bounds)
      __builtin_prefetch(Kbh + (size_t)(k0 + 16 + lo) * ATT_D, 0, 3);
    v8f c = qk_tile(a0, a1, Kbh, k0, lo, half);
#pragma unroll
    for (int v = 0; v < 8; ++v) {
      const bool msk = (k0 + lo) > (q0 + v + 8 * half);
      const float s  = msk ? -__builtin_inff() : c[v] * CSC;
      const float mn = fmaxf(m[v], s);
      const float e  = msk ? 0.0f : fexp2(s - mn);
      sm[v] = ((m[v] == mn) ? sm[v] : sm[v] * fexp2(m[v] - mn)) + e;
      m[v]  = mn;
    }
  }
  // Reduce (m, sum) across the 16 lanes of each half (rows live across lanes).
#pragma unroll
  for (int d = 1; d < 16; d <<= 1) {
#pragma unroll
    for (int v = 0; v < 8; ++v) {
      const float mo = __shfl_xor(m[v], d, 32);
      const float so = __shfl_xor(sm[v], d, 32);
      const float mn = fmaxf(m[v], mo);
      sm[v] = ((m[v] == mn) ? sm[v] : sm[v] * fexp2(m[v] - mn)) +
              ((mo  == mn) ? so    : so    * fexp2(mo  - mn));
      m[v] = mn;
    }
  }
  float inv[8];
#pragma unroll
  for (int v = 0; v < 8; ++v) inv[v] = 1.0f / sm[v];

  // ---------------- Pass B: weights write + P*V accumulate ------------------
  __shared__ __align__(16) float Pbuf[16][36];  // 16x32 P tile pair (+pad)

  v8f o[4];
#pragma unroll
  for (int t = 0; t < 4; ++t) o[t] = (v8f){};

  const f32x4 z4 = {0.f, 0.f, 0.f, 0.f};

  for (int p = 0; p < 64; ++p) {           // pairs of 16-wide k tiles
    const int k0p = p << 5;
    if (k0p <= q0 + 15) {
      // --- compute normalized P for both 16-col tiles into LDS (C-layout) ---
#pragma unroll
      for (int jj = 0; jj < 2; ++jj) {
        const int k0 = k0p + (jj << 4);
        if (k0 <= q0 + 15) {
          v8f c = qk_tile(a0, a1, Kbh, k0, lo, half);
#pragma unroll
          for (int v = 0; v < 8; ++v) {
            const bool msk = (k0 + lo) > (q0 + v + 8 * half);
            const float pr = msk ? 0.0f : fexp2(c[v] * CSC - m[v]) * inv[v];
            Pbuf[v + 8 * half][(jj << 4) + lo] = pr;
          }
        } else {
#pragma unroll
          for (int v = 0; v < 8; ++v) Pbuf[v + 8 * half][(jj << 4) + lo] = 0.0f;
        }
      }
      __syncthreads();

      // --- stream weights out (16 rows x 32 cols), coalesced NT float4 ------
#pragma unroll
      for (int it = 0; it < 4; ++it) {
        const int fid = it * 32 + lane;
        const int row = fid >> 3;
        const int c4  = (fid & 7) << 2;
        const f32x4 w = *(const f32x4*)&Pbuf[row][c4];
        __builtin_nontemporal_store(
            w, (f32x4*)(Wbh + (size_t)(q0 + row) * ATT_S + k0p + c4));
      }

      // --- reload P as A-matrix (16x32) with f32->f16 conversion ------------
      v16h pa;
#pragma unroll
      for (int i = 0; i < 16; ++i) {
        const int k = (i < 8) ? (8 * half + i) : (16 + 8 * half + (i - 8));
        pa[i] = (_Float16)Pbuf[lo][k];
      }
      __syncthreads();

      // --- O += P(16x32) * V(32x64): 4 WMMAs, one per 16-wide d chunk -------
#pragma unroll
      for (int t = 0; t < 4; ++t) {
        v16h bv;
#pragma unroll
        for (int i = 0; i < 16; ++i)
          bv[i] = (_Float16)Vbh[(size_t)(k0p + ((i < 8) ? (8 * half + i) : (16 + 8 * half + (i - 8)))) * ATT_D + 16 * t + lo];
        o[t] = __builtin_amdgcn_wmma_f32_16x16x32_f16(false, pa, false, bv,
                                                      (short)0, o[t], false, false);
      }
    } else {
      // fully masked pair: weights are zeros (still must be written)
#pragma unroll
      for (int it = 0; it < 4; ++it) {
        const int fid = it * 32 + lane;
        const int row = fid >> 3;
        const int c4  = (fid & 7) << 2;
        __builtin_nontemporal_store(
            z4, (f32x4*)(Wbh + (size_t)(q0 + row) * ATT_S + k0p + c4));
      }
    }
  }

  // ---------------- store O tile (rows M = v+8*half, cols 16t+lo) ------------
#pragma unroll
  for (int t = 0; t < 4; ++t) {
#pragma unroll
    for (int v = 0; v < 8; ++v) {
      __builtin_nontemporal_store(
          o[t][v], Obh + (size_t)(q0 + v + 8 * half) * ATT_D + 16 * t + lo);
    }
  }
}

extern "C" void kernel_launch(void* const* d_in, const int* in_sizes, int n_in,
                              void* d_out, int out_size, void* d_ws, size_t ws_size,
                              hipStream_t stream) {
  const float* Q = (const float*)d_in[0];
  const float* K = (const float*)d_in[1];
  const float* V = (const float*)d_in[2];
  // d_in[3] = mask (unused: causal mask applied analytically)

  float* O = (float*)d_out;                                   // [B,H,S,D]
  float* W = O + (size_t)ATT_B * ATT_H * ATT_S * ATT_D;       // [B,H,S,S]

  const int blocks = ATT_B * ATT_H * (ATT_S / 16);            // 4096
  attn_fwd_kernel<<<dim3(blocks), dim3(32), 0, stream>>>(Q, K, V, O, W);
}
